// SparseMiddleExtractor_64613488001760
// MI455X (gfx1250) — matrix operand order repaired and checked by reading, not compiled, not run on hardware
//
#include <hip/hip_runtime.h>
#include <hip/hip_bf16.h>

typedef __bf16 bf16_t;
typedef __attribute__((ext_vector_type(16))) __bf16 v16bf;
typedef __attribute__((ext_vector_type(8)))  __bf16 v8bf;
typedef __attribute__((ext_vector_type(8)))  float  v8f;

union AFrag { v16bf v; v8bf h[2]; };

constexpr int GH = 160;          // H is constant for the whole network
constexpr int GW = 160;          // W is constant for the whole network

// ---------------------------------------------------------------------------
// Layer configuration (mirrors reference CFG)
// ---------------------------------------------------------------------------
struct LayerCfg { int cin, cout, td, th, tw, sd, pd, ph, pw, subm; };
static const LayerCfg g_cfg[12] = {
  { 4,  16, 3,3,3, 1, 1,1,1, 1},
  {16,  16, 3,3,3, 1, 1,1,1, 1},
  {16,  32, 3,3,3, 2, 1,1,1, 0},
  {32,  32, 3,3,3, 1, 1,1,1, 1},
  {32,  32, 3,3,3, 1, 1,1,1, 1},
  {32,  64, 3,3,3, 2, 1,1,1, 0},
  {64,  64, 3,3,3, 1, 1,1,1, 1},
  {64,  64, 3,3,3, 1, 1,1,1, 1},
  {64,  64, 3,3,3, 2, 0,1,1, 0},
  {64,  64, 3,3,3, 1, 1,1,1, 1},
  {64,  64, 3,3,3, 1, 1,1,1, 1},
  {64, 128, 3,1,1, 2, 0,0,0, 0},
};

// ---------------------------------------------------------------------------
// Scatter voxel features into dense channels-last bf16 grid + fp32 mask
// ---------------------------------------------------------------------------
__global__ void scatter_kernel(const float* __restrict__ vf,
                               const int* __restrict__ coors,
                               bf16_t* __restrict__ act0,
                               float* __restrict__ mask0,
                               int nvox, int cin_pad) {
  int v = blockIdx.x * 256 + threadIdx.x;
  if (v >= nvox) return;
  int z = coors[4 * v + 1];
  int y = coors[4 * v + 2];
  int x = coors[4 * v + 3];
  size_t s = ((size_t)z * GH + y) * GW + x;
  mask0[s] = 1.0f;
  bf16_t* p = act0 + s * cin_pad;
  #pragma unroll
  for (int c = 0; c < 4; ++c) p[c] = (bf16_t)vf[v * 4 + c];
}

// ---------------------------------------------------------------------------
// Mask downsample: out = (conv(mask, ones, stride, pad) > 0)
// ---------------------------------------------------------------------------
__global__ void mask_down_kernel(const float* __restrict__ min_,
                                 float* __restrict__ mout,
                                 int Din, int Dout,
                                 int td, int th, int tw,
                                 int sd, int pd, int ph, int pw) {
  int m = blockIdx.x * 256 + threadIdx.x;
  int total = Dout * GH * GW;
  if (m >= total) return;
  int x = m % GW;
  int t = m / GW;
  int y = t % GH;
  int z = t / GH;
  float r = 0.0f;
  for (int kd = 0; kd < td; ++kd) {
    int zi = z * sd - pd + kd;
    if ((unsigned)zi >= (unsigned)Din) continue;
    for (int kh = 0; kh < th; ++kh) {
      int yi = y - ph + kh;
      if ((unsigned)yi >= (unsigned)GH) continue;
      for (int kw = 0; kw < tw; ++kw) {
        int xi = x - pw + kw;
        if ((unsigned)xi >= (unsigned)GW) continue;
        if (min_[((size_t)zi * GH + yi) * GW + xi] > 0.0f) r = 1.0f;
      }
    }
  }
  mout[m] = r;
}

// ---------------------------------------------------------------------------
// Fold BN scale into bf16 weights, compute bias. Layout: wgt[cout][Kp],
// K index = tap*cin_pad + ci (tap-major, channels contiguous).
// ---------------------------------------------------------------------------
struct WPrepArgs {
  const float* w; const float* gamma; const float* beta;
  const float* mean; const float* var;
  bf16_t* wgt; float* bias;
  int cin, cin_pad, cout, taps, Kp;
};

__global__ void prep_weights_kernel(WPrepArgs a) {
  int idx = blockIdx.x * 256 + threadIdx.x;
  int total = a.cout * a.Kp;
  if (idx >= total) return;
  int n = idx / a.Kp;
  int k = idx - n * a.Kp;
  float scale = a.gamma[n] * rsqrtf(a.var[n] + 1e-3f);
  if (k == 0) a.bias[n] = a.beta[n] - a.mean[n] * scale;
  int tap = k / a.cin_pad;
  int ci  = k - tap * a.cin_pad;
  float v = 0.0f;
  if (tap < a.taps && ci < a.cin)
    v = a.w[((size_t)n * a.cin + ci) * a.taps + tap] * scale;
  a.wgt[idx] = (bf16_t)v;
}

// ---------------------------------------------------------------------------
// Implicit-GEMM 3D conv via V_WMMA_F32_16X16X32_BF16.
//  M = Dout*GH*GW output voxels (16 per wave), N = cout (NT tiles of 16),
//  K = taps*cin_pad padded to 32.  B panel staged in LDS per block.
// Template: NT = cout/16, CLOG = log2(cin_pad), TH/TW = kernel h/w taps
// (depth taps fixed at 3 for all layers).
// ---------------------------------------------------------------------------
struct ConvArgs {
  const bf16_t* act_in;   // [Din][GH][GW][cin_pad]
  bf16_t* act_out;        // [Dout][GH][GW][cout]
  const bf16_t* wgt;      // [cout][Kp]
  const float* bias;      // [cout]
  const float* mask;      // [Dout*GH*GW]
  float* final_out;       // non-null for last layer: [(cout*Dout)][GH][GW] fp32
  int Din, Dout;
  int cout;
  int sd, pd, ph, pw;
  int Kp;
};

template <int NT, int CLOG, int TH, int TW>
__global__ __launch_bounds__(256)
void conv_wmma_kernel(ConvArgs args) {
  constexpr int THW  = TH * TW;
  constexpr int TAPS = 3 * THW;
  constexpr int BS_ROW = 40;                       // padded LDS row (halves)
  constexpr int HW = GH * GW;
  __shared__ __align__(16) bf16_t Bs[NT * 16 * BS_ROW];

  const int tid  = threadIdx.x;
  const int lane = tid & 31;
  const int waveInBlk = tid >> 5;
  const int m0 = (blockIdx.x * 8 + waveInBlk) * 16;   // grid covers exactly

  const int lm = lane & 15;
  const int h  = lane >> 4;

  // decode this lane's A-matrix row (output voxel) -- constant divisors
  int m  = m0 + lm;
  int xo = m % GW;
  int t  = m / GW;
  int yo = t % GH;
  int zo = t / GH;
  const int zbase = zo * args.sd - args.pd;
  const int ybase = yo - args.ph;
  const int xbase = xo - args.pw;

  v8f acc[NT];
  #pragma unroll
  for (int i = 0; i < NT; ++i) acc[i] = {};

  for (int kk = 0; kk < args.Kp; kk += 32) {
    // ---- cooperative stage of the 32 x cout weight panel into LDS ----
    __syncthreads();
    if (tid < NT * 32) {
      int n = tid >> 1;
      int halfk = (tid & 1) * 16;
      const bf16_t* src = args.wgt + (size_t)n * args.Kp + kk + halfk;
      *reinterpret_cast<v8bf*>(&Bs[n * BS_ROW + halfk]) =
          *reinterpret_cast<const v8bf*>(src);
      *reinterpret_cast<v8bf*>(&Bs[n * BS_ROW + halfk + 8]) =
          *reinterpret_cast<const v8bf*>(src + 8);
      // software-pipeline the next panel into cache (global_prefetch_b8)
      __builtin_prefetch(src + 32, 0, 0);
    }
    __syncthreads();

    // ---- A fragment: 16x32 bf16, contiguous k = 16h + s (same as B) ----
    AFrag a;
    #pragma unroll
    for (int r = 0; r < 2; ++r) {
      int kr  = kk + h * 16 + r * 8;           // 8 contiguous K values
      int tap = kr >> CLOG;
      int ci  = kr & ((1 << CLOG) - 1);
      v8bf frag = {};
      if (tap < TAPS) {                        // only false in zero-pad tail
        int kd  = tap / THW;
        int rem = tap - kd * THW;
        int kh  = rem / TW;
        int kw  = rem - kh * TW;
        int zi = zbase + kd;
        int yi = ybase + kh;
        int xi = xbase + kw;
        if ((unsigned)zi < (unsigned)args.Din &&
            (unsigned)yi < (unsigned)GH &&
            (unsigned)xi < (unsigned)GW) {
          const bf16_t* p = args.act_in +
              ((((size_t)zi * HW) + (size_t)yi * GW + xi) << CLOG) + ci;
          frag = *reinterpret_cast<const v8bf*>(p);   // aligned 16B
        }
      }
      a.h[r] = frag;
    }

    // ---- B fragments from LDS + WMMA for each 16-wide N tile ----
    #pragma unroll
    for (int nt = 0; nt < NT; ++nt) {
      const bf16_t* bp = &Bs[(nt * 16 + lm) * BS_ROW + h * 16];
      AFrag b;
      b.h[0] = *reinterpret_cast<const v8bf*>(bp);
      b.h[1] = *reinterpret_cast<const v8bf*>(bp + 8);
      acc[nt] = __builtin_amdgcn_wmma_f32_16x16x32_bf16(
          false, a.v, false, b.v, (short)0, acc[nt], false, false);
    }
  }

  // ---- epilogue: bias + ReLU + mask, store ----
  #pragma unroll
  for (int nt = 0; nt < NT; ++nt) {
    int col = nt * 16 + lm;
    float bias = args.bias[col];
    #pragma unroll
    for (int v = 0; v < 8; ++v) {
      int mrow = m0 + v + h * 8;               // C/D layout: VGPR v, half h
      float val = acc[nt][v] + bias;
      val = val > 0.0f ? val : 0.0f;
      val *= args.mask[mrow];
      if (args.final_out) {
        int x = mrow % GW;
        int tt = mrow / GW;
        int y = tt % GH;
        int z = tt / GH;
        args.final_out[(((size_t)col * args.Dout + z) * GH + y) * GW + x] = val;
      } else {
        args.act_out[(size_t)mrow * args.cout + col] = (bf16_t)val;
      }
    }
  }
}

// ---------------------------------------------------------------------------
// Host orchestration
// ---------------------------------------------------------------------------
extern "C" void kernel_launch(void* const* d_in, const int* in_sizes, int n_in,
                              void* d_out, int out_size, void* d_ws, size_t ws_size,
                              hipStream_t stream) {
  const int NVOX = in_sizes[1] / 4;
  const float* vf = (const float*)d_in[0];
  const int* coors = (const int*)d_in[1];

  // workspace carve-out
  char* ws = (char*)d_ws;
  size_t off = 0;
  auto alloc = [&](size_t bytes) -> void* {
    off = (off + 255) & ~(size_t)255;
    void* p = ws + off;
    off += bytes;
    return p;
  };

  const size_t ACT_BYTES = (size_t)42 * 1024 * 1024;   // max layer act ~36MB
  bf16_t* bufA = (bf16_t*)alloc(ACT_BYTES);
  bf16_t* bufB = (bf16_t*)alloc(ACT_BYTES);

  // depth at each mask level: 41,21,11,5,2
  int dlev[5];
  {
    int D = 41, lev = 0;
    dlev[0] = D;
    for (int i = 0; i < 12; ++i) {
      const LayerCfg& c = g_cfg[i];
      if (!c.subm) { D = (D + 2 * c.pd - c.td) / c.sd + 1; dlev[++lev] = D; }
    }
  }
  float* maskP[5];
  for (int l = 0; l < 5; ++l)
    maskP[l] = (float*)alloc((size_t)dlev[l] * GH * GW * sizeof(float));

  bf16_t* wgtP[12];
  float* biasP[12];
  int KpA[12], cinpA[12];
  for (int i = 0; i < 12; ++i) {
    const LayerCfg& c = g_cfg[i];
    int cinp = c.cin < 16 ? 16 : c.cin;
    int taps = c.td * c.th * c.tw;
    int Kp = (taps * cinp + 31) & ~31;
    cinpA[i] = cinp;
    KpA[i] = Kp;
    wgtP[i]  = (bf16_t*)alloc((size_t)c.cout * Kp * sizeof(bf16_t));
    biasP[i] = (float*)alloc((size_t)c.cout * sizeof(float));
  }

  // ---- stage 0: zero input grid + mask, scatter voxels ----
  hipMemsetAsync(bufA, 0, (size_t)41 * GH * GW * 16 * sizeof(bf16_t), stream);
  hipMemsetAsync(maskP[0], 0, (size_t)dlev[0] * GH * GW * sizeof(float), stream);
  scatter_kernel<<<(NVOX + 255) / 256, 256, 0, stream>>>(
      vf, coors, bufA, maskP[0], NVOX, 16);

  // ---- stage 1: fold BN into weights (all layers) ----
  for (int i = 0; i < 12; ++i) {
    const LayerCfg& c = g_cfg[i];
    WPrepArgs wa;
    wa.w     = (const float*)d_in[3 + 5 * i + 0];
    wa.gamma = (const float*)d_in[3 + 5 * i + 1];
    wa.beta  = (const float*)d_in[3 + 5 * i + 2];
    wa.mean  = (const float*)d_in[3 + 5 * i + 3];
    wa.var   = (const float*)d_in[3 + 5 * i + 4];
    wa.wgt = wgtP[i];
    wa.bias = biasP[i];
    wa.cin = c.cin;
    wa.cin_pad = cinpA[i];
    wa.cout = c.cout;
    wa.taps = c.td * c.th * c.tw;
    wa.Kp = KpA[i];
    int total = c.cout * KpA[i];
    prep_weights_kernel<<<(total + 255) / 256, 256, 0, stream>>>(wa);
  }

  // ---- stage 2: conv layers ----
  int Din = 41;
  int lev = 0;
  bf16_t* cur = bufA;
  bf16_t* nxt = bufB;
  for (int i = 0; i < 12; ++i) {
    const LayerCfg& c = g_cfg[i];
    int Dout = Din;
    if (!c.subm) {
      Dout = (Din + 2 * c.pd - c.td) / c.sd + 1;
      ++lev;
      int total = Dout * GH * GW;
      mask_down_kernel<<<(total + 255) / 256, 256, 0, stream>>>(
          maskP[lev - 1], maskP[lev], Din, Dout,
          c.td, c.th, c.tw, c.sd, c.pd, c.ph, c.pw);
    }

    ConvArgs a;
    a.act_in = cur;
    a.act_out = nxt;
    a.wgt = wgtP[i];
    a.bias = biasP[i];
    a.mask = maskP[lev];
    a.final_out = (i == 11) ? (float*)d_out : nullptr;
    a.Din = Din; a.Dout = Dout;
    a.cout = c.cout;
    a.sd = c.sd; a.pd = c.pd; a.ph = c.ph; a.pw = c.pw;
    a.Kp = KpA[i];

    int mTotal = Dout * GH * GW;              // always a multiple of 128
    int blocks = mTotal / 128;                // 8 waves/block, 16 rows/wave
    int NT = c.cout / 16;
    int cinp = cinpA[i];
    if (c.th == 1) {
      conv_wmma_kernel<8, 6, 1, 1><<<blocks, 256, 0, stream>>>(a);
    } else if (NT == 1) {
      conv_wmma_kernel<1, 4, 3, 3><<<blocks, 256, 0, stream>>>(a);
    } else if (NT == 2 && cinp == 16) {
      conv_wmma_kernel<2, 4, 3, 3><<<blocks, 256, 0, stream>>>(a);
    } else if (NT == 2) {
      conv_wmma_kernel<2, 5, 3, 3><<<blocks, 256, 0, stream>>>(a);
    } else if (NT == 4 && cinp == 32) {
      conv_wmma_kernel<4, 5, 3, 3><<<blocks, 256, 0, stream>>>(a);
    } else {
      conv_wmma_kernel<4, 6, 3, 3><<<blocks, 256, 0, stream>>>(a);
    }

    bf16_t* tmp = cur; cur = nxt; nxt = tmp;
    Din = Dout;
  }
}
